// SlidingWindowEncoder_66735201845494
// MI455X (gfx1250) — matrix operand order repaired and checked by reading, not compiled, not run on hardware
//
#include <hip/hip_runtime.h>

typedef __attribute__((ext_vector_type(16))) __bf16 v16bf;
typedef __attribute__((ext_vector_type(8)))  float  v8f;
typedef unsigned int u32x4 __attribute__((ext_vector_type(4)));
typedef int          i32x8 __attribute__((ext_vector_type(8)));
typedef int          i32x4 __attribute__((ext_vector_type(4)));

#define WIN_   256
#define NHEAD  8
#define DMODEL 256
#define DHEAD  32
#define DFF_   1024
#define NROWS  65536        // B * S = 8 * 8192
#define EPSLN  1e-5f

#ifndef __has_builtin
#define __has_builtin(x) 0
#endif
#if defined(__HIP_DEVICE_COMPILE__) && __has_builtin(__builtin_amdgcn_tensor_load_to_lds)
#define HAVE_TDM 1
#else
#define HAVE_TDM 0
#endif
#if defined(__has_include)
#if __has_include(<hip/amd_detail/amd_gfx1250_TDM.h>)
#define TDM_ARITY6 1
#else
#define TDM_ARITY6 0
#endif
#else
#define TDM_ARITY6 0
#endif

// ---------------------------------------------------------------------------
// WMMA helpers (CDNA5 16x16x32 bf16 -> f32)
// ---------------------------------------------------------------------------
static __device__ inline v8f wmma_bf16(v16bf a, v16bf b, v8f c) {
  // 8 args: (neg_a, A, neg_b, B, c_mod, C, reuse_a, reuse_b)
  return __builtin_amdgcn_wmma_f32_16x16x32_bf16(false, a, false, b,
                                                 (short)0, c, false, false);
}

static __device__ inline v8f zero8() {
  v8f z = {0.f, 0.f, 0.f, 0.f, 0.f, 0.f, 0.f, 0.f};
  return z;
}

// A fragment: 16x32 (MxK) bf16. lane: m = lane&15, half = lane>>4.
// element j (0..15): k = k0 + half*16 + j
static __device__ inline v16bf load_a_frag(const __bf16* p, int row0, int ld,
                                           int k0, int lane) {
  const int m = lane & 15, h = lane >> 4;
  const __bf16* q = p + (row0 + m) * ld + k0 + h * 16;
  v16bf a;
#pragma unroll
  for (int j = 0; j < 16; ++j) a[j] = q[j];
  return a;
}

// B fragment: 32x16 (KxN) bf16 stored row-major with leading dim ld.
// lane: n = n0 + (lane&15), half = lane>>4; element j: k = k0 + half*16 + j
static __device__ inline v16bf load_b_frag(const __bf16* p, int k0, int n0,
                                           int ld, int lane) {
  const int n = n0 + (lane & 15), h = lane >> 4;
  const __bf16* q = p + (k0 + h * 16) * ld + n;
  v16bf b;
#pragma unroll
  for (int j = 0; j < 16; ++j) b[j] = q[j * ld];
  return b;
}

// B fragment for K^T: B[k][n] = Kh[n][k], Kh is 256x32 row-major.
// lane reads 16 *contiguous* bf16 from row n (k = half*16 + j).
static __device__ inline v16bf load_bT_frag(const __bf16* Kh, int n0, int lane) {
  const int n = n0 + (lane & 15), h = lane >> 4;
  const __bf16* q = Kh + n * DHEAD + h * 16;
  v16bf b;
#pragma unroll
  for (int j = 0; j < 16; ++j) b[j] = q[j];
  return b;
}

// ---------------------------------------------------------------------------
// Tensor Data Mover: async 2D bf16 tile load global -> LDS.
// D# bit layout per CDNA5 ISA.
// LDS offset = low 32 bits of generic pointer (LDS aperture, ISA §10.2).
// ---------------------------------------------------------------------------
#if HAVE_TDM
static __device__ inline unsigned lds_off(const void* p) {
  return (unsigned)(unsigned long long)(uintptr_t)p;
}

static __device__ inline void tdm_load_2d_bf16(unsigned lds_addr, const void* gptr,
                                               unsigned tile_w, unsigned tile_h,
                                               unsigned stride_elems) {
  const unsigned long long ga = (unsigned long long)(uintptr_t)gptr;
  u32x4 g0;
  g0[0] = 1u;                                   // count=1, user mode, no gather
  g0[1] = lds_addr;                             // lds_addr (bytes)
  g0[2] = (unsigned)ga;                         // global_addr[31:0]
  g0[3] = (unsigned)((ga >> 32) & 0x01FFFFFFu)  // global_addr[56:32]
          | (2u << 30);                         // type = 2 ("image")
  i32x8 g1;
  g1[0] = (int)(1u << 16);                      // mask=0, data_size=1 (2 bytes)
  g1[1] = (int)(tile_w << 16);                  // tensor_dim0[15:0] (== tile_w)
  g1[2] = (int)((tile_w >> 16) | (tile_h << 16));  // tdim0 hi | tensor_dim1 lo
  g1[3] = (int)((tile_h >> 16) | (tile_w << 16));  // tdim1 hi | tile_dim0
  g1[4] = (int)(tile_h & 0xFFFFu);              // tile_dim1 | tile_dim2=0
  g1[5] = (int)stride_elems;                    // tensor_dim0_stride[31:0]
  g1[6] = 0;                                    // stride hi | dim1_stride lo
  g1[7] = 0;
  i32x4 z4 = {0, 0, 0, 0};
#if TDM_ARITY6
  i32x8 z8 = {0, 0, 0, 0, 0, 0, 0, 0};
  __builtin_amdgcn_tensor_load_to_lds(g0, g1, z4, z4, z8, 0);
#else
  __builtin_amdgcn_tensor_load_to_lds(g0, g1, z4, z4, 0);
#endif
}
#endif  // HAVE_TDM

// ---------------------------------------------------------------------------
// Generic WMMA GEMM:  Out = act(A @ W + bias)
// A: M x K (f32 or bf16), W: K x N f32 (converted to bf16 in registers)
// Block = 256 threads (8 waves), tile 128x128, BK = 32, double-buffered LDS.
// Pipeline per k-step: commit regs/TDM(buf cur) -> wait TENSORcnt -> barrier
// -> issue TDM + global loads for k+1 (buf nxt) -> WMMA on buf cur.
// Only ONE barrier per iteration: iteration i+1 writes the opposite buffer,
// and its barrier is reached only after every wave finished compute(i).
// ---------------------------------------------------------------------------
template <bool A_IS_BF16, bool RELU, bool OUT_BF16>
__global__ __launch_bounds__(256) void gemm_bias_kernel(
    const void* __restrict__ Av, const float* __restrict__ W,
    const float* __restrict__ bias, void* __restrict__ Outv,
    int M, int N, int K) {
  __shared__ __bf16 As[2][128 * 32];
  __shared__ __bf16 Bs[2][32 * 128];

  const int tid = threadIdx.x;
  const int lane = tid & 31;
  const int w = tid >> 5;
  const int wr = w >> 1, wc = w & 1;
  const int m0 = blockIdx.y * 128;
  const int n0 = blockIdx.x * 128;
  const int nk = K >> 5;

  // W staging: each thread owns 4 rows (stride 8) x 4 consecutive cols
  const int w_row = tid >> 5;          // 0..7
  const int w_col = (tid & 31) * 4;    // 0..124
  // f32-A staging: each thread owns 4 rows (stride 32) x 4 consecutive cols
  const int a_row = tid >> 3;          // 0..31
  const int a_col = (tid & 7) * 4;     // 0..28

  v8f acc[2][4];
#pragma unroll
  for (int t = 0; t < 2; ++t)
#pragma unroll
    for (int c = 0; c < 4; ++c) acc[t][c] = zero8();

  float4 wreg[4];
  float4 areg[4];

  auto load_w = [&](int k0) {
#pragma unroll
    for (int j = 0; j < 4; ++j)
      wreg[j] = *(const float4*)(W + (size_t)(k0 + w_row + j * 8) * N + n0 + w_col);
  };
  auto load_a32 = [&](int k0) {
    const float* A = (const float*)Av;
#pragma unroll
    for (int j = 0; j < 4; ++j)
      areg[j] = *(const float4*)(A + (size_t)(m0 + a_row + j * 32) * K + k0 + a_col);
  };

  // ---- prologue: stage k=0 ----
  load_w(0);
  if (A_IS_BF16) {
#if HAVE_TDM
    if (w == 0)
      tdm_load_2d_bf16(lds_off(&As[0][0]), (const __bf16*)Av + (size_t)m0 * K,
                       32, 128, (unsigned)K);
#endif
  } else {
    load_a32(0);
  }

  for (int i = 0; i < nk; ++i) {
    const int cur = i & 1;
    const int k0 = i * 32;

    // ---- commit staged data for step i into LDS buffer `cur` ----
#pragma unroll
    for (int j = 0; j < 4; ++j) {
      __bf16* d = &Bs[cur][(w_row + j * 8) * 128 + w_col];
      d[0] = (__bf16)wreg[j].x; d[1] = (__bf16)wreg[j].y;
      d[2] = (__bf16)wreg[j].z; d[3] = (__bf16)wreg[j].w;
    }
    if (A_IS_BF16) {
#if HAVE_TDM
      if (w == 0) __builtin_amdgcn_s_wait_tensorcnt(0);
#else
      const __bf16* A = (const __bf16*)Av;
#pragma unroll
      for (int j = 0; j < 2; ++j) {
        int idx = tid + j * 256;             // 512 chunks of 8 bf16
        int r = idx >> 2, c8 = (idx & 3) * 8;
        *(uint4*)(&As[cur][r * 32 + c8]) =
            *(const uint4*)(A + (size_t)(m0 + r) * K + k0 + c8);
      }
#endif
    } else {
#pragma unroll
      for (int j = 0; j < 4; ++j) {
        __bf16* d = &As[cur][(a_row + j * 32) * 32 + a_col];
        d[0] = (__bf16)areg[j].x; d[1] = (__bf16)areg[j].y;
        d[2] = (__bf16)areg[j].z; d[3] = (__bf16)areg[j].w;
      }
    }
    __syncthreads();

    // ---- kick off stage i+1 into the other buffer (overlaps WMMA below) ----
    if (i + 1 < nk) {
      load_w(k0 + 32);
      if (A_IS_BF16) {
#if HAVE_TDM
        if (w == 0)
          tdm_load_2d_bf16(lds_off(&As[cur ^ 1][0]),
                           (const __bf16*)Av + (size_t)m0 * K + k0 + 32,
                           32, 128, (unsigned)K);
#endif
      } else {
        load_a32(k0 + 32);
      }
      if (i + 2 < nk)  // L2 prefetch two steps ahead -> global_prefetch_b8
        __builtin_prefetch(W + (size_t)(k0 + 64 + (tid >> 3)) * N + n0 + (tid & 7) * 16, 0, 3);
    }

    // ---- WMMA on buffer `cur` ----
    v16bf a0f = load_a_frag(&As[cur][0], wr * 32, 32, 0, lane);
    v16bf a1f = load_a_frag(&As[cur][0], wr * 32 + 16, 32, 0, lane);
#pragma unroll
    for (int c = 0; c < 4; ++c) {
      v16bf b = load_b_frag(&Bs[cur][0], 0, wc * 64 + c * 16, 128, lane);
      acc[0][c] = wmma_bf16(a0f, b, acc[0][c]);
      acc[1][c] = wmma_bf16(a1f, b, acc[1][c]);
    }
    // no trailing barrier: next iteration writes the opposite buffer, and its
    // top barrier is only passed once every wave finished this compute.
  }

  // ---- epilogue: bias (+ReLU), store bf16 or f32 ----
  const int h = lane >> 4, nl = lane & 15;
#pragma unroll
  for (int t = 0; t < 2; ++t) {
#pragma unroll
    for (int c = 0; c < 4; ++c) {
      const int col = n0 + wc * 64 + c * 16 + nl;
      const float bv = bias[col];
#pragma unroll
      for (int r = 0; r < 8; ++r) {
        const int row = m0 + wr * 32 + t * 16 + r + 8 * h;
        float v = acc[t][c][r] + bv;
        if (RELU) v = fmaxf(v, 0.f);
        if (OUT_BF16)
          ((__bf16*)Outv)[(size_t)row * N + col] = (__bf16)v;
        else
          ((float*)Outv)[(size_t)row * N + col] = v;
      }
    }
  }
}

// ---------------------------------------------------------------------------
// Windowed multi-head attention: one block per (window, head).
// LDS: Qh/Kh/Vh (256x32 bf16 each) + P (256x256 bf16) = 176 KB (CDNA5 320KB).
// 8 waves; wave w owns query rows w*32..w*32+31.
// Q/K/V head slices are staged by the Tensor Data Mover (waves 0/1/2 issue).
// ---------------------------------------------------------------------------
__global__ __launch_bounds__(256) void attn_kernel(
    const __bf16* __restrict__ Q, const __bf16* __restrict__ Kw,
    const __bf16* __restrict__ Vw, __bf16* __restrict__ Ctx) {
  extern __shared__ char smem[];
  __bf16* Qh = (__bf16*)smem;            // 256*32
  __bf16* Kh = Qh + WIN_ * DHEAD;
  __bf16* Vh = Kh + WIN_ * DHEAD;
  __bf16* P  = Vh + WIN_ * DHEAD;        // 256*256

  const int tid = threadIdx.x, lane = tid & 31, w = tid >> 5;
  const int win = blockIdx.x >> 3, hd = blockIdx.x & 7;
  const size_t rowbase = (size_t)win * WIN_;
  const int coff = hd * DHEAD;
  const int h = lane >> 4, nl = lane & 15;

#if HAVE_TDM
  // one async 256x32 tile per wave 0/1/2; stride = DMODEL elements
  if (w < 3) {
    const __bf16* src = (w == 0) ? Q : (w == 1) ? Kw : Vw;
    __bf16* dst = (w == 0) ? Qh : (w == 1) ? Kh : Vh;
    tdm_load_2d_bf16(lds_off(dst), src + rowbase * DMODEL + coff,
                     DHEAD, WIN_, DMODEL);
    __builtin_amdgcn_s_wait_tensorcnt(0);
  }
#else
#pragma unroll
  for (int i = 0; i < 4; ++i) {
    int idx = tid + i * 256;
    int r = idx >> 2, c = (idx & 3) * 8;
    const size_t g = (rowbase + r) * DMODEL + coff + c;
    *(uint4*)(Qh + r * DHEAD + c) = *(const uint4*)(Q + g);
    *(uint4*)(Kh + r * DHEAD + c) = *(const uint4*)(Kw + g);
    *(uint4*)(Vh + r * DHEAD + c) = *(const uint4*)(Vw + g);
  }
#endif
  __syncthreads();

  const float scale = 0.1767766952966369f;  // 1/sqrt(32)

  v16bf aq0 = load_a_frag(Qh, w * 32, DHEAD, 0, lane);
  v16bf aq1 = load_a_frag(Qh, w * 32 + 16, DHEAD, 0, lane);

  // ---- pass 1: row maxima of raw scores ----
  float mx[2][8];
#pragma unroll
  for (int t = 0; t < 2; ++t)
#pragma unroll
    for (int r = 0; r < 8; ++r) mx[t][r] = -3.0e38f;

  for (int ct = 0; ct < 16; ++ct) {
    v16bf bk = load_bT_frag(Kh, ct * 16, lane);
    v8f s0 = wmma_bf16(aq0, bk, zero8());
    v8f s1 = wmma_bf16(aq1, bk, zero8());
#pragma unroll
    for (int r = 0; r < 8; ++r) {
      mx[0][r] = fmaxf(mx[0][r], s0[r]);
      mx[1][r] = fmaxf(mx[1][r], s1[r]);
    }
  }
#pragma unroll
  for (int m = 1; m < 16; m <<= 1) {   // reduce within 16-lane half (same row)
#pragma unroll
    for (int t = 0; t < 2; ++t)
#pragma unroll
      for (int r = 0; r < 8; ++r)
        mx[t][r] = fmaxf(mx[t][r], __shfl_xor(mx[t][r], m));
  }

  // ---- pass 2: exp(scale*(s-mx)) -> P (bf16, unnormalized), track sums ----
  float sm[2][8];
#pragma unroll
  for (int t = 0; t < 2; ++t)
#pragma unroll
    for (int r = 0; r < 8; ++r) sm[t][r] = 0.f;

  for (int ct = 0; ct < 16; ++ct) {
    v16bf bk = load_bT_frag(Kh, ct * 16, lane);
    v8f s0 = wmma_bf16(aq0, bk, zero8());
    v8f s1 = wmma_bf16(aq1, bk, zero8());
#pragma unroll
    for (int r = 0; r < 8; ++r) {
      float e0 = __expf((s0[r] - mx[0][r]) * scale);
      float e1 = __expf((s1[r] - mx[1][r]) * scale);
      sm[0][r] += e0;
      sm[1][r] += e1;
      P[(w * 32 + r + 8 * h) * WIN_ + ct * 16 + nl]      = (__bf16)e0;
      P[(w * 32 + 16 + r + 8 * h) * WIN_ + ct * 16 + nl] = (__bf16)e1;
    }
  }
  float rinv[2][8];
#pragma unroll
  for (int m = 1; m < 16; m <<= 1)
#pragma unroll
    for (int t = 0; t < 2; ++t)
#pragma unroll
      for (int r = 0; r < 8; ++r) sm[t][r] += __shfl_xor(sm[t][r], m);
#pragma unroll
  for (int t = 0; t < 2; ++t)
#pragma unroll
    for (int r = 0; r < 8; ++r) rinv[t][r] = 1.0f / sm[t][r];

  // ---- ctx = P @ V  (each wave reads only its own 32 rows of P) ----
  v8f acc[2][2] = {{zero8(), zero8()}, {zero8(), zero8()}};
  for (int ks = 0; ks < 16; ++ks) {
    const int k0 = ks * 32;
    v16bf a0 = load_a_frag(P, w * 32, WIN_, k0, lane);
    v16bf a1 = load_a_frag(P, w * 32 + 16, WIN_, k0, lane);
#pragma unroll
    for (int nt = 0; nt < 2; ++nt) {
      v16bf bv = load_b_frag(Vh, k0, nt * 16, DHEAD, lane);
      acc[0][nt] = wmma_bf16(a0, bv, acc[0][nt]);
      acc[1][nt] = wmma_bf16(a1, bv, acc[1][nt]);
    }
  }

#pragma unroll
  for (int t = 0; t < 2; ++t)
#pragma unroll
    for (int nt = 0; nt < 2; ++nt)
#pragma unroll
      for (int r = 0; r < 8; ++r) {
        const size_t row = rowbase + w * 32 + t * 16 + r + 8 * h;
        const int col = coff + nt * 16 + nl;
        Ctx[row * DMODEL + col] = (__bf16)(acc[t][nt][r] * rinv[t][r]);
      }
}

// ---------------------------------------------------------------------------
// Residual + LayerNorm over 256-wide rows. One wave per row (8 rows / block).
// ---------------------------------------------------------------------------
template <bool WRITE_BF16>
__global__ __launch_bounds__(256) void residual_ln_kernel(
    const float* __restrict__ A, const float* __restrict__ Bv,
    const float* __restrict__ g, const float* __restrict__ bt,
    float* __restrict__ Out, __bf16* __restrict__ Outb) {
  const int row = blockIdx.x * 8 + (threadIdx.x >> 5);
  const int lane = threadIdx.x & 31;
  const size_t base = (size_t)row * DMODEL + lane * 8;

  float4 a0 = *(const float4*)(A + base);
  float4 a1 = *(const float4*)(A + base + 4);
  float4 b0 = *(const float4*)(Bv + base);
  float4 b1 = *(const float4*)(Bv + base + 4);
  float v[8] = {a0.x + b0.x, a0.y + b0.y, a0.z + b0.z, a0.w + b0.w,
                a1.x + b1.x, a1.y + b1.y, a1.z + b1.z, a1.w + b1.w};

  float s = 0.f;
#pragma unroll
  for (int i = 0; i < 8; ++i) s += v[i];
#pragma unroll
  for (int m = 1; m < 32; m <<= 1) s += __shfl_xor(s, m);
  const float mu = s * (1.f / 256.f);

  float q = 0.f;
#pragma unroll
  for (int i = 0; i < 8; ++i) { float d = v[i] - mu; q += d * d; }
#pragma unroll
  for (int m = 1; m < 32; m <<= 1) q += __shfl_xor(q, m);
  const float rstd = rsqrtf(q * (1.f / 256.f) + EPSLN);

#pragma unroll
  for (int i = 0; i < 8; ++i) {
    const int c = lane * 8 + i;
    const float y = (v[i] - mu) * rstd * g[c] + bt[c];
    Out[(size_t)row * DMODEL + c] = y;
    if (WRITE_BF16) Outb[(size_t)row * DMODEL + c] = (__bf16)y;
  }
}

// ---------------------------------------------------------------------------
// Host orchestration
// ---------------------------------------------------------------------------
extern "C" void kernel_launch(void* const* d_in, const int* in_sizes, int n_in,
                              void* d_out, int out_size, void* d_ws,
                              size_t ws_size, hipStream_t stream) {
  (void)in_sizes; (void)n_in; (void)out_size; (void)ws_size;
  const float* q_in  = (const float*)d_in[0];
  const float* kv_in = (const float*)d_in[1];
  const float* Wq = (const float*)d_in[2];  const float* bq = (const float*)d_in[3];
  const float* Wk = (const float*)d_in[4];  const float* bk = (const float*)d_in[5];
  const float* Wv = (const float*)d_in[6];  const float* bv = (const float*)d_in[7];
  const float* Wo = (const float*)d_in[8];  const float* bo = (const float*)d_in[9];
  const float* ln1_g = (const float*)d_in[10]; const float* ln1_b = (const float*)d_in[11];
  const float* W1 = (const float*)d_in[12]; const float* b1 = (const float*)d_in[13];
  const float* W2 = (const float*)d_in[14]; const float* b2 = (const float*)d_in[15];
  const float* ln2_g = (const float*)d_in[16]; const float* ln2_b = (const float*)d_in[17];

  // workspace layout (bytes)
  char* ws = (char*)d_ws;
  const size_t SZ_BF = (size_t)NROWS * DMODEL * 2;   // 33.5 MB
  const size_t SZ_F  = (size_t)NROWS * DMODEL * 4;   // 67.1 MB
  __bf16* Qb   = (__bf16*)(ws);
  __bf16* Kb   = (__bf16*)(ws + SZ_BF);
  __bf16* Vb   = (__bf16*)(ws + 2 * SZ_BF);
  __bf16* Ctxb = (__bf16*)(ws + 3 * SZ_BF);
  float*  aproj = (float*)(ws + 4 * SZ_BF);
  float*  xf    = (float*)(ws + 4 * SZ_BF + SZ_F);
  __bf16* xb    = (__bf16*)(ws + 4 * SZ_BF + 2 * SZ_F);
  __bf16* hid   = (__bf16*)(ws + 5 * SZ_BF + 2 * SZ_F);
  float*  ff2   = (float*)(ws + 5 * SZ_BF + 2 * SZ_F + (size_t)NROWS * DFF_ * 2);

  const dim3 blk(256);
  const dim3 gD(DMODEL / 128, NROWS / 128);   // (2, 512)
  const dim3 gF(DFF_ / 128, NROWS / 128);     // (8, 512)

  // 1) Q/K/V projections (f32 in -> bf16 out)
  gemm_bias_kernel<false, false, true><<<gD, blk, 0, stream>>>(
      q_in, Wq, bq, Qb, NROWS, DMODEL, DMODEL);
  gemm_bias_kernel<false, false, true><<<gD, blk, 0, stream>>>(
      kv_in, Wk, bk, Kb, NROWS, DMODEL, DMODEL);
  gemm_bias_kernel<false, false, true><<<gD, blk, 0, stream>>>(
      kv_in, Wv, bv, Vb, NROWS, DMODEL, DMODEL);

  // 2) windowed attention, one block per (window, head); 176 KB dynamic LDS
  const int attn_lds = (3 * WIN_ * DHEAD + WIN_ * WIN_) * 2;  // 180224 B
  hipFuncSetAttribute((const void*)attn_kernel,
                      hipFuncAttributeMaxDynamicSharedMemorySize, attn_lds);
  attn_kernel<<<dim3(256 * NHEAD), blk, attn_lds, stream>>>(Qb, Kb, Vb, Ctxb);

  // 3) output projection (bf16 in -> f32 out)
  gemm_bias_kernel<true, false, false><<<gD, blk, 0, stream>>>(
      Ctxb, Wo, bo, aproj, NROWS, DMODEL, DMODEL);

  // 4) x = LN(q_in + attn_out); emit f32 + bf16
  residual_ln_kernel<true><<<dim3(NROWS / 8), blk, 0, stream>>>(
      q_in, aproj, ln1_g, ln1_b, xf, xb);

  // 5) FFN layer 1 with ReLU (bf16 in -> bf16 out)
  gemm_bias_kernel<true, true, true><<<gF, blk, 0, stream>>>(
      xb, W1, b1, hid, NROWS, DFF_, DMODEL);

  // 6) FFN layer 2 (bf16 in -> f32 out)
  gemm_bias_kernel<true, false, false><<<gD, blk, 0, stream>>>(
      hid, W2, b2, ff2, NROWS, DMODEL, DFF_);

  // 7) y = LN(x + ff) -> d_out (f32)
  residual_ln_kernel<false><<<dim3(NROWS / 8), blk, 0, stream>>>(
      xf, ff2, ln2_g, ln2_b, (float*)d_out, nullptr);
}
